// Model_83141976916220
// MI455X (gfx1250) — matrix operand order repaired and checked by reading, not compiled, not run on hardware
//
#include <hip/hip_runtime.h>

// ---------------------------------------------------------------------------
// Problem constants (from reference): B=4, T=2048, C=768, H=12, hs=64
// ---------------------------------------------------------------------------
#define BATCH 4
#define TSEQ  2048
#define CDIM  768
#define N3    2304          // 3*C
#define NHEAD 12
#define HS    64
#define BT    8192          // BATCH*TSEQ
#define BH    48            // BATCH*NHEAD

typedef __attribute__((ext_vector_type(16))) __bf16 v16bf;
typedef __attribute__((ext_vector_type(8)))  float  v8f;

union Frag {                 // 32 bytes: one bf16 WMMA A/B operand per lane
  v16bf v;
  uint4 q[2];
};

__device__ __forceinline__ unsigned short f32_to_bf16(float f) {
  unsigned int u = __float_as_uint(f);
  // round-to-nearest-even (data is finite/normal-ish; no NaN special case)
  u += 0x7FFFu + ((u >> 16) & 1u);
  return (unsigned short)(u >> 16);
}

__device__ __forceinline__ v8f wmma_bf16(v16bf a, v16bf b, v8f c) {
  // (neg_a, A, neg_b, B, c_mod, C, reuse_a, reuse_b)
  return __builtin_amdgcn_wmma_f32_16x16x32_bf16(false, a, false, b, (short)0, c,
                                                 false, false);
}

// ---------------------------------------------------------------------------
// Kernel 0a: fp32 x -> bf16 (same row-major layout, (B*T, C))
// ---------------------------------------------------------------------------
__global__ void cvt_x_kernel(const float* __restrict__ x,
                             unsigned short* __restrict__ xb, int n) {
  int i = blockIdx.x * blockDim.x + threadIdx.x;
  if (i < n) xb[i] = f32_to_bf16(x[i]);
}

// ---------------------------------------------------------------------------
// Kernel 0b: W (C, 3C) fp32 -> Wt (3C, C) bf16  (transpose so WMMA B-fragments
// read 32 contiguous bytes per lane along K)
// ---------------------------------------------------------------------------
__global__ void cvt_w_kernel(const float* __restrict__ W,
                             unsigned short* __restrict__ Wt) {
  int i = blockIdx.x * blockDim.x + threadIdx.x;
  if (i < CDIM * N3) {
    int k = i / N3;
    int n = i % N3;
    Wt[(size_t)n * CDIM + k] = f32_to_bf16(W[i]);
  }
}

// ---------------------------------------------------------------------------
// Kernel 1: QKV projection. One 16(M) x 64(N) output tile per wave:
// each A-fragment load is reused by 4 WMMA against 4 W column-tiles.
// A 64-wide N block maps to exactly one (q/k/v, head) pair (768 % 64 == 0).
//   Qb, Kb : (B,H,T,64) bf16 row-major
//   Vt     : (B,H,64,T) bf16 (transposed for att@V B-fragments)
// Fragment layouts as loaded here (A and B share one contiguous k order):
//   A 16x32 bf16: lane<16 -> row=lane,     elems K0..15 (contiguous)
//                 lane>=16 -> row=lane-16, elems K16..31
//   B 32x16 bf16: lane<16 -> col=lane,     elems K0..15 (contiguous)
//                 lane>=16 -> col=lane-16, elems K16..31
//   D 16x16 f32:  lane = N + 16*(M>=8), vgpr r -> M = r + 8*(lane>>4)
// ---------------------------------------------------------------------------
__global__ void qkv_gemm_kernel(const unsigned short* __restrict__ xb,
                                const unsigned short* __restrict__ Wt,
                                const float* __restrict__ bias,
                                unsigned short* __restrict__ Qb,
                                unsigned short* __restrict__ Kb,
                                unsigned short* __restrict__ Vt) {
  const int wave = blockIdx.x * (blockDim.x >> 5) + (threadIdx.x >> 5);
  const int NT = N3 / 64;                 // 36 column blocks
  const int mt = wave / NT;               // 0..511 row tile (over B*T)
  const int nt = wave % NT;               // 0..35  column block (over 3C)
  const int lane = threadIdx.x & 31;
  const int half = lane >> 4;
  const int l15  = lane & 15;

  const unsigned short* arow = xb + (size_t)(mt * 16 + l15) * CDIM;
  const int nbase = nt * 64;

  v8f acc[4] = {};
  for (int k0 = 0; k0 < CDIM; k0 += 32) {
    Frag a;
    const unsigned short* ap = arow + k0 + half * 16;
    a.q[0] = *(const uint4*)(ap);
    a.q[1] = *(const uint4*)(ap + 8);
#pragma unroll
    for (int s = 0; s < 4; ++s) {
      Frag b;
      const unsigned short* bp =
          Wt + (size_t)(nbase + s * 16 + l15) * CDIM + k0 + half * 16;
      b.q[0] = *(const uint4*)(bp);
      b.q[1] = *(const uint4*)(bp + 8);
      acc[s] = wmma_bf16(a.v, b.v, acc[s]);
    }
  }

  // Column block -> (which, head) is uniform; d = s*16 + l15.
  const int which = nbase / CDIM;         // 0=Q 1=K 2=V
  const int h     = (nbase % CDIM) >> 6;  // head
  const int mbase = mt * 16 + half * 8;   // this lane's 8 output rows
  const int b_    = mbase / TSEQ;         // rows never straddle a batch

  unsigned short* dstQK = (which == 0) ? Qb : Kb;
#pragma unroll
  for (int s = 0; s < 4; ++s) {
    const int d  = s * 16 + l15;
    const float bv = bias[nbase + d];
#pragma unroll
    for (int r = 0; r < 8; ++r) {
      const int m = mbase + r;
      const int t = m & (TSEQ - 1);
      const unsigned short w = f32_to_bf16(acc[s][r] + bv);
      if (which < 2) {
        dstQK[((size_t)(b_ * NHEAD + h) * TSEQ + t) * HS + d] = w;
      } else {
        Vt[((size_t)(b_ * NHEAD + h) * HS + d) * TSEQ + t] = w;
      }
    }
  }
}

// ---------------------------------------------------------------------------
// Kernel 2: causal ReLU attention. One wave per (b,h,32-query strip).
// Per 32-key step: 8 WMMA for scores (2 q-tiles x 2 key-halves x K=64),
// scale/mask/ReLU in f32, LDS bounce to re-layout D(lane=key)->A(lane=query),
// then 8 WMMA for att@V — K and V fragments are shared by both q-tiles,
// doubling FLOP/byte vs a 16-query strip.
// Single-wave workgroup => __syncthreads() is a free compiler fence (S_NOP).
// ---------------------------------------------------------------------------
__global__ void attn_kernel(const unsigned short* __restrict__ Qb,
                            const unsigned short* __restrict__ Kb,
                            const unsigned short* __restrict__ Vt,
                            float* __restrict__ out) {
  __shared__ __align__(16) unsigned short sS[2][16 * 32];  // 2 KB score staging

  const int tile = blockIdx.x;        // BH * (T/32) = 3072
  const int qt   = tile & 63;
  const int bh   = tile >> 6;
  const int qs   = qt << 5;           // base of this wave's 32 queries
  const int lane = threadIdx.x & 31;
  const int half = lane >> 4;
  const int l15  = lane & 15;

  const unsigned short* Q = Qb + (size_t)bh * TSEQ * HS;
  const unsigned short* K = Kb + (size_t)bh * TSEQ * HS;
  const unsigned short* V = Vt + (size_t)bh * HS * TSEQ;

  // Q A-fragments: 2 query tiles x 2 hs-K ranges ([0,32) and [32,64))
  Frag aq[2][2];
#pragma unroll
  for (int g = 0; g < 2; ++g) {
    const unsigned short* qrow = Q + (size_t)(qs + g * 16 + l15) * HS + half * 16;
    aq[g][0].q[0] = *(const uint4*)(qrow);
    aq[g][0].q[1] = *(const uint4*)(qrow + 8);
    aq[g][1].q[0] = *(const uint4*)(qrow + 32);
    aq[g][1].q[1] = *(const uint4*)(qrow + 40);
  }

  v8f y[2][4] = {};                   // 32 queries x 64 hs accumulator

  const int kend = qs + 32;           // causal: keys <= qs+31 needed
  for (int kt = 0; kt < kend; kt += 32) {
    // Prefetch next key-step's K rows and V cachelines (global_prefetch_b8)
    if (kt + 32 < kend) {
      __builtin_prefetch(K + (size_t)(kt + 32 + lane) * HS, 0, 1);
      __builtin_prefetch(V + (size_t)(2 * lane) * TSEQ + kt + 32, 0, 1);
      __builtin_prefetch(V + (size_t)(2 * lane + 1) * TSEQ + kt + 32, 0, 1);
    }

#pragma unroll
    for (int s = 0; s < 2; ++s) {
      const int kb = kt + s * 16;
      // K^T B-fragments: lane = key col, elems = hs dims (contiguous in Kb row)
      const unsigned short* krow = K + (size_t)(kb + l15) * HS + half * 16;
      Frag b0, b1;
      b0.q[0] = *(const uint4*)(krow);
      b0.q[1] = *(const uint4*)(krow + 8);
      b1.q[0] = *(const uint4*)(krow + 32);
      b1.q[1] = *(const uint4*)(krow + 40);

      const int kj = kb + l15;        // this lane's key column
#pragma unroll
      for (int g = 0; g < 2; ++g) {
        v8f sacc = {};
        sacc = wmma_bf16(aq[g][0].v, b0.v, sacc);
        sacc = wmma_bf16(aq[g][1].v, b1.v, sacc);
        // scale + causal mask + ReLU, pack to bf16 into 16x32 LDS tile g
#pragma unroll
        for (int r = 0; r < 8; ++r) {
          const int qi = qs + g * 16 + half * 8 + r;
          float sv = sacc[r] * 0.125f;     // 1/sqrt(64)
          sv = (kj <= qi && sv > 0.0f) ? sv : 0.0f;
          sS[g][(half * 8 + r) * 32 + s * 16 + l15] = f32_to_bf16(sv);
        }
      }
    }
    __syncthreads();                  // fence: LDS writes -> cross-lane reads

    // Reload scores as A-fragments (16 queries x 32 keys, per q-tile)
    Frag as0, as1;
    {
      const unsigned short* s0 = sS[0] + l15 * 32 + half * 16;
      const unsigned short* s1 = sS[1] + l15 * 32 + half * 16;
      as0.q[0] = *(const uint4*)(s0);
      as0.q[1] = *(const uint4*)(s0 + 8);
      as1.q[0] = *(const uint4*)(s1);
      as1.q[1] = *(const uint4*)(s1 + 8);
    }
    __syncthreads();                  // fence before next iteration overwrites

    // att @ V over the 4 hs column blocks; each V B-fragment feeds both
    // q-tiles. V is (64,T) so lanes read contiguous key runs per hs column.
#pragma unroll
    for (int t4 = 0; t4 < 4; ++t4) {
      const unsigned short* vrow =
          V + (size_t)(t4 * 16 + l15) * TSEQ + kt + half * 16;
      Frag bv;
      bv.q[0] = *(const uint4*)(vrow);
      bv.q[1] = *(const uint4*)(vrow + 8);
      y[0][t4] = wmma_bf16(as0.v, bv.v, y[0][t4]);
      y[1][t4] = wmma_bf16(as1.v, bv.v, y[1][t4]);
    }
  }

  // Store y: (B,T,C) fp32; lanes 0..15 of a half-wave write 64 contiguous bytes
  const int b_ = bh / NHEAD;
  const int h  = bh % NHEAD;
#pragma unroll
  for (int g = 0; g < 2; ++g) {
#pragma unroll
    for (int t4 = 0; t4 < 4; ++t4) {
#pragma unroll
      for (int r = 0; r < 8; ++r) {
        const int t = qs + g * 16 + half * 8 + r;
        out[(size_t)(b_ * TSEQ + t) * CDIM + h * HS + t4 * 16 + l15] =
            y[g][t4][r];
      }
    }
  }
}

// ---------------------------------------------------------------------------
// Host launcher
// ---------------------------------------------------------------------------
extern "C" void kernel_launch(void* const* d_in, const int* in_sizes, int n_in,
                              void* d_out, int out_size, void* d_ws,
                              size_t ws_size, hipStream_t stream) {
  const float* x    = (const float*)d_in[0];   // (B,T,C)
  const float* W    = (const float*)d_in[1];   // (C,3C)
  const float* bias = (const float*)d_in[2];   // (3C,)
  float* out        = (float*)d_out;

  char* ws = (char*)d_ws;
  const size_t XB_BYTES   = (size_t)BT * CDIM * 2;        // 12,582,912
  const size_t WT_BYTES   = (size_t)N3 * CDIM * 2;        //  3,538,944
  const size_t HEAD_BYTES = (size_t)BH * TSEQ * HS * 2;   // 12,582,912 each

  unsigned short* xb = (unsigned short*)(ws);
  unsigned short* Wt = (unsigned short*)(ws + XB_BYTES);
  unsigned short* Qb = (unsigned short*)(ws + XB_BYTES + WT_BYTES);
  unsigned short* Kb = (unsigned short*)(ws + XB_BYTES + WT_BYTES + HEAD_BYTES);
  unsigned short* Vt =
      (unsigned short*)(ws + XB_BYTES + WT_BYTES + 2 * HEAD_BYTES);

  // 0) fp32 -> bf16 conversions (+ W transpose)
  {
    const int n = BT * CDIM;                       // 6,291,456
    cvt_x_kernel<<<(n + 255) / 256, 256, 0, stream>>>(x, xb, n);
  }
  {
    const int n = CDIM * N3;                       // 1,769,472
    cvt_w_kernel<<<(n + 255) / 256, 256, 0, stream>>>(W, Wt);
  }

  // 1) QKV projection: 512 x 36 = 18432 wave-tiles (16x64 each),
  //    4 waves per 128-thread block
  {
    const int waves = (BT / 16) * (N3 / 64);
    qkv_gemm_kernel<<<waves / 4, 128, 0, stream>>>(xb, Wt, bias, Qb, Kb, Vt);
  }

  // 2) causal ReLU attention: one wave per (b,h,32-query strip)
  {
    const int tiles = BH * (TSEQ / 32);            // 3072
    attn_kernel<<<tiles, 32, 0, stream>>>(Qb, Kb, Vt, out);
  }
}
